// ct_loss_50517405335943
// MI455X (gfx1250) — matrix operand order repaired and checked by reading, not compiled, not run on hardware
//
#include <hip/hip_runtime.h>

typedef __attribute__((ext_vector_type(16))) __bf16 v16bf;
typedef __attribute__((ext_vector_type(8)))  __bf16 v8bf;
typedef __attribute__((ext_vector_type(8)))  float  v8f;

#define D_DIM 1024
#define C_CLS 1000
#define C_PAD 1008
#define XSTR  1032   // padded LDS row stride in elements (2064B: 16B-aligned, bank-skewed)

__global__ void k_zero_out(float* out) {
  if (threadIdx.x == 0) out[0] = 0.0f;
}

// Convert prototypes fp32 -> bf16, zero-pad rows [1000,1008)
__global__ __launch_bounds__(128) void k_conv_pad_p(const float* __restrict__ p,
                                                    __bf16* __restrict__ pb) {
  const int c = blockIdx.x;          // 0..C_PAD-1
  const int t = threadIdx.x;         // 128 threads * 8 elems = 1024
  __bf16* dst = pb + (size_t)c * D_DIM + t * 8;
  if (c < C_CLS) {
    v8f v = *(const v8f*)(p + (size_t)c * D_DIM + t * 8);
    *(v8bf*)dst = __builtin_convertvector(v, v8bf);
  } else {
    v8bf z = {};
    *(v8bf*)dst = z;
  }
}

// Fused: normalize 16 rows -> LDS bf16, bf16 WMMA GEMM vs all 1000 prototypes,
// online log-softmax + label gather, atomic accumulate mean loss.
__global__ __launch_bounds__(256) void k_gemm_ce(
    const float* __restrict__ x, const __bf16* __restrict__ pb,
    const int* __restrict__ labels, float* __restrict__ out, float inv_b) {
  __shared__ __bf16 lds_x[16 * XSTR];
  __shared__ float lds_m[8][16];
  __shared__ float lds_s[8][16];
  __shared__ float lds_lab[16];
  __shared__ float lds_loss[16];
  __shared__ int   lds_label[16];

  const int t    = threadIdx.x;
  const int wave = t >> 5;
  const int lane = t & 31;
  const int half = lane >> 4;
  const int l15  = lane & 15;
  const size_t row0 = (size_t)blockIdx.x * 16;
  const float NEG_INF = -__builtin_inff();

  // ---- stage: load 16 x-rows once, L2-normalize, store bf16 to LDS ----
  {
    const int r  = t >> 4;        // 0..15: row within tile
    const int cs = (t & 15) * 64; // 64-column segment per thread
    const v8f* src = (const v8f*)(x + (row0 + r) * D_DIM + cs);
    v8f buf[8];
    float ss = 0.0f;
#pragma unroll
    for (int i = 0; i < 8; ++i) {
      buf[i] = src[i];
#pragma unroll
      for (int j = 0; j < 8; ++j) ss += buf[i][j] * buf[i][j];
    }
    // the 16 threads of one row occupy one 16-lane half-wave
#pragma unroll
    for (int off = 1; off < 16; off <<= 1) ss += __shfl_xor(ss, off, 32);
    const float inv = rsqrtf(ss);
    __bf16* dst = lds_x + r * XSTR + cs;
#pragma unroll
    for (int i = 0; i < 8; ++i)
      *(v8bf*)(dst + i * 8) = __builtin_convertvector(buf[i] * inv, v8bf);
  }
  if (t < 16) {
    lds_lab[t]   = 0.0f;
    lds_label[t] = labels[row0 + t];
  }
  __syncthreads();

  // per-lane online-softmax state for the 8 rows (half*8 + v) this half owns
  float m[8], s[8];
#pragma unroll
  for (int v = 0; v < 8; ++v) { m[v] = NEG_INF; s[v] = 0.0f; }

  // A fragment source: lane M = l15, K chunks at half*8 and 16+half*8
  const __bf16* aprow = lds_x + l15 * XSTR + half * 8;

  for (int tile = wave; tile < C_PAD / 16; tile += 8) {
    const int col = tile * 16 + l15;
    // B fragment source: lane N = l15 (prototype row 'col'), K chunks at half*8 and 16+half*8
    const __bf16* bptr = pb + (size_t)col * D_DIM + half * 8;
    v8f acc = {};
#pragma unroll
    for (int k = 0; k < D_DIM; k += 32) {
      v8bf a0 = *(const v8bf*)(aprow + k);        // K = k +     half*8 + 0..7
      v8bf a1 = *(const v8bf*)(aprow + k + 16);   // K = k + 16 + half*8 + 0..7
      v8bf b0 = *(const v8bf*)(bptr + k);         // K = k +      half*8 + 0..7
      v8bf b1 = *(const v8bf*)(bptr + k + 16);    // K = k + 16 + half*8 + 0..7
      v16bf a = __builtin_shufflevector(a0, a1, 0,1,2,3,4,5,6,7,8,9,10,11,12,13,14,15);
      v16bf b = __builtin_shufflevector(b0, b1, 0,1,2,3,4,5,6,7,8,9,10,11,12,13,14,15);
      acc = __builtin_amdgcn_wmma_f32_16x16x32_bf16(false, a, false, b, (short)0, acc,
                                                    false, false);
    }
    // ---- tile epilogue: online softmax over this 16-col slab ----
    const bool valid = (col < C_CLS);
#pragma unroll
    for (int v = 0; v < 8; ++v) {
      const int rrow = half * 8 + v;              // D-matrix row = vgpr + half*8
      float val = valid ? acc[v] : NEG_INF;
      if (valid && col == lds_label[rrow]) lds_lab[rrow] = acc[v];
      float tmax = val;
#pragma unroll
      for (int off = 1; off < 16; off <<= 1)
        tmax = fmaxf(tmax, __shfl_xor(tmax, off, 32));
      const float nm = fmaxf(m[v], tmax);
      float e = __expf(val - nm);
#pragma unroll
      for (int off = 1; off < 16; off <<= 1) e += __shfl_xor(e, off, 32);
      s[v] = s[v] * __expf(m[v] - nm) + e;
      m[v] = nm;
    }
  }

  // publish per-wave (m,s): lanes 0 and 16 hold rows 0-7 / 8-15
  if (l15 == 0) {
#pragma unroll
    for (int v = 0; v < 8; ++v) {
      lds_m[wave][half * 8 + v] = m[v];
      lds_s[wave][half * 8 + v] = s[v];
    }
  }
  __syncthreads();

  // merge 8 waves' partial softmax per row, form per-row loss
  if (t < 16) {
    float M = NEG_INF;
#pragma unroll
    for (int w = 0; w < 8; ++w) M = fmaxf(M, lds_m[w][t]);
    float S = 0.0f;
#pragma unroll
    for (int w = 0; w < 8; ++w) S += lds_s[w][t] * __expf(lds_m[w][t] - M);
    lds_loss[t] = (M + __logf(S)) - lds_lab[t];   // -(logit_y - logsumexp)
  }
  __syncthreads();
  if (t == 0) {
    float sum = 0.0f;
#pragma unroll
    for (int r = 0; r < 16; ++r) sum += lds_loss[r];
    atomicAdd(out, sum * inv_b);
  }
}

extern "C" void kernel_launch(void* const* d_in, const int* in_sizes, int n_in,
                              void* d_out, int out_size, void* d_ws, size_t ws_size,
                              hipStream_t stream) {
  (void)n_in; (void)out_size; (void)ws_size;
  const float* x  = (const float*)d_in[0];   // [B, 1024] fp32
  const float* p  = (const float*)d_in[1];   // [1000, 1024] fp32
  const int*   lb = (const int*)d_in[2];     // [B] labels
  float* out = (float*)d_out;                // [1] fp32

  const int B = in_sizes[0] / D_DIM;         // 16384
  __bf16* pb = (__bf16*)d_ws;                // 1008*1024 bf16 = ~2.06 MB scratch

  k_zero_out<<<1, 64, 0, stream>>>(out);
  k_conv_pad_p<<<C_PAD, 128, 0, stream>>>(p, pb);
  k_gemm_ce<<<B / 16, 256, 0, stream>>>(x, pb, lb, out, 1.0f / (float)B);
}
